// CrossRepair_63462436765733
// MI455X (gfx1250) — matrix operand order repaired and checked by reading, not compiled, not run on hardware
//
#include <hip/hip_runtime.h>
#include <hip/hip_bf16.h>
#include <math.h>

// ---------- types ----------
typedef __bf16        v16bf __attribute__((ext_vector_type(16)));
typedef float         v8f   __attribute__((ext_vector_type(8)));
typedef unsigned int  u32x4 __attribute__((ext_vector_type(4)));
typedef int           i32x4 __attribute__((ext_vector_type(4)));
typedef int           i32x8 __attribute__((ext_vector_type(8)));

union Frag { u32x4 u[2]; v16bf v; };

static __device__ __forceinline__ unsigned short f32_to_bf16(float f) {
    unsigned int u = __float_as_uint(f);
    unsigned int r = (u + 0x7FFFu + ((u >> 16) & 1u)) >> 16;   // round-to-nearest-even
    return (unsigned short)r;
}

// Problem constants
#define BB 2
#define TT 2048
#define DD 1024
#define HH 16
#define DHH 64
#define MM (BB * TT)      // 4096
#define D4 (4 * DD)       // 4096

// ---------- TDM: 2D tile global -> LDS, bf16 elements, padded LDS rows ----------
// Copies tile_d1 rows of tile_d0 bf16 elements; source rows are row_stride_elems
// apart; destination rows land at (tile_d0*2 + 16)-byte stride in LDS via the
// TDM pad feature (4 DWORDs after every 32 DWORDs).
static __device__ __forceinline__ void tdm_load_tile(const unsigned short* gsrc,
                                                     unsigned lds_off,
                                                     int tile_d0, int tile_d1,
                                                     long long row_stride_elems) {
    unsigned long long ga = (unsigned long long)(uintptr_t)gsrc;
    unsigned long long st = (unsigned long long)row_stride_elems;  // in 2B units
    i32x4 g0;
    g0[0] = 1;                                            // count=1, user descriptor
    g0[1] = (int)lds_off;                                 // lds_addr (bytes)
    g0[2] = (int)(ga & 0xffffffffull);                    // global_addr[31:0]
    g0[3] = (int)((ga >> 32) & 0x1ffffffull) | (int)(2u << 30);  // addr[56:32] | type=2
    i32x8 g1;
    g1[0] = (1 << 16)            // data_size = 2 bytes
          | (1 << 20)            // pad_enable
          | (4 << 22)            // pad_interval: every 32 DWORDs (=64 bf16)
          | (3 << 25);           // pad_amount: 4 DWORDs (=16 bytes)
    g1[1] = (tile_d0 & 0xffff) << 16;                                  // tensor_dim0[15:0]
    g1[2] = (int)(((unsigned)tile_d0 >> 16) | ((unsigned)(tile_d1 & 0xffff) << 16));
    g1[3] = (int)(((unsigned)tile_d1 >> 16) | ((unsigned)(tile_d0 & 0xffff) << 16)); // | tile_dim0
    g1[4] = (tile_d1 & 0xffff);                                        // tile_dim1 (tile_dim2=0)
    g1[5] = (int)(st & 0xffffffffull);                                 // dim0_stride[31:0]
    g1[6] = (int)((st >> 32) & 0xffffull);                             // dim0_stride[47:32]
    g1[7] = 0;
    asm volatile("tensor_load_to_lds %0, %1" :: "s"(g0), "s"(g1) : "memory");
}

// ---------- weight transpose + fp32 -> bf16 convert ----------
// in: (K x N) row-major fp32; out: (N x K) row-major bf16
__global__ void k_transpose_bf16(const float* __restrict__ in,
                                 unsigned short* __restrict__ out,
                                 int K, int N) {
    __shared__ float tile[32][33];
    int n0 = blockIdx.x * 32, k0 = blockIdx.y * 32;
    int tx = threadIdx.x, ty = threadIdx.y;          // block (32,8)
    for (int i = ty; i < 32; i += 8)
        tile[i][tx] = in[(size_t)(k0 + i) * N + (n0 + tx)];
    __syncthreads();
    for (int i = ty; i < 32; i += 8)
        out[(size_t)(n0 + i) * K + (k0 + tx)] = f32_to_bf16(tile[tx][i]);
}

// ---------- LayerNorm (one wave per 1024-wide row) -> bf16 ----------
__global__ void k_layernorm_bf16(const float* __restrict__ x,
                                 const float* __restrict__ g,
                                 const float* __restrict__ bta,
                                 unsigned short* __restrict__ out,
                                 int rows) {
    int wid  = (blockIdx.x * blockDim.x + threadIdx.x) >> 5;
    int lane = threadIdx.x & 31;
    if (wid >= rows) return;
    const float* xr = x + (size_t)wid * DD;
    float vals[DD / 32];
    float s = 0.f;
#pragma unroll
    for (int j = 0; j < DD / 32; ++j) { vals[j] = xr[lane + 32 * j]; s += vals[j]; }
#pragma unroll
    for (int off = 16; off; off >>= 1) s += __shfl_xor(s, off, 32);
    float mean = s * (1.0f / DD);
    float s2 = 0.f;
#pragma unroll
    for (int j = 0; j < DD / 32; ++j) { float d = vals[j] - mean; s2 += d * d; }
#pragma unroll
    for (int off = 16; off; off >>= 1) s2 += __shfl_xor(s2, off, 32);
    float rstd = rsqrtf(s2 * (1.0f / DD) + 1e-5f);
    unsigned short* orow = out + (size_t)wid * DD;
#pragma unroll
    for (int j = 0; j < DD / 32; ++j) {
        int c = lane + 32 * j;
        orow[c] = f32_to_bf16((vals[j] - mean) * rstd * g[c] + bta[c]);
    }
}

// ---------- WMMA GEMM:  C[M,N] = A[M,K](bf16) * W^T[N,K](bf16) + bias ----------
// Block = 8 waves, tile 128(M) x 64(N), K staged 64 at a time into LDS by the
// Tensor Data Mover (double-buffered), WMMAs feed from LDS (ds_load_b128).
// All fragments of a k-step are hoisted before the WMMA burst so the DS loads
// issue as one clause and retire with staggered s_wait_dscnt.
// EPI: 0 = bias only, 1 = bias + exact GELU, 2 = bias + residual
// OUTB: 1 -> bf16 output (Cb), 0 -> fp32 output (Cf)
template <int EPI, int OUTB>
__global__ void __launch_bounds__(256)
k_gemm(const unsigned short* __restrict__ A,
       const unsigned short* __restrict__ WT,
       const float* __restrict__ bias,
       const float* __restrict__ resid,
       float* __restrict__ Cf,
       unsigned short* __restrict__ Cb,
       int M, int N, int K) {
    __shared__ unsigned short ldsA[2][128 * 72];   // 72 = 64 data + 8 pad (TDM pad)
    __shared__ unsigned short ldsB[2][64 * 72];

    int tid  = threadIdx.x;
    int wav  = tid >> 5;
    int lane = tid & 31;
    int r    = lane & 15;
    int half = lane >> 4;

    int nbm = M >> 7;                  // M/128
    int bm  = blockIdx.x % nbm;
    int bn  = blockIdx.x / nbm;        // over N/64

    const unsigned short* Ablk = A  + (size_t)bm * 128 * K;
    const unsigned short* Bblk = WT + (size_t)bn * 64  * K;

    unsigned ldsA0 = (unsigned)(uintptr_t)(void*)&ldsA[0][0];
    unsigned ldsA1 = (unsigned)(uintptr_t)(void*)&ldsA[1][0];
    unsigned ldsB0 = (unsigned)(uintptr_t)(void*)&ldsB[0][0];
    unsigned ldsB1 = (unsigned)(uintptr_t)(void*)&ldsB[1][0];

    v8f acc[4] = {};

    int niter = K >> 6;                // K/64 chunks
    if (wav == 0) {                    // prologue: fill buffer 0
        tdm_load_tile(Ablk, ldsA0, 64, 128, K);
        tdm_load_tile(Bblk, ldsB0, 64, 64, K);
    }
    for (int it = 0; it < niter; ++it) {
        int cur = it & 1;
        if (wav == 0) {
            if (it + 1 < niter) {      // prefetch next chunk into other buffer
                int kt = (it + 1) << 6;
                tdm_load_tile(Ablk + kt, cur ? ldsA0 : ldsA1, 64, 128, K);
                tdm_load_tile(Bblk + kt, cur ? ldsB0 : ldsB1, 64, 64, K);
                __builtin_amdgcn_s_wait_tensorcnt(2);  // current buffer's 2 TDMs done
            } else {
                __builtin_amdgcn_s_wait_tensorcnt(0);
            }
        }
        __syncthreads();               // publish current buffer

        const unsigned short* ap  = &ldsA[cur][(wav * 16 + r) * 72 + half * 8];
        const unsigned short* bp0 = &ldsB[cur][r * 72 + half * 8];
#pragma unroll
        for (int ks = 0; ks < 2; ++ks) {
            int kl = ks * 32;
            Frag fa, fb[4];
            // hoist all loads of this k-step: one DS clause, staggered waits
            fa.u[0] = *(const u32x4*)(ap + kl);        // K = kl+8h .. +7
            fa.u[1] = *(const u32x4*)(ap + kl + 16);   // K = kl+16+8h .. +7
#pragma unroll
            for (int nt = 0; nt < 4; ++nt) {
                const unsigned short* bp = bp0 + nt * 16 * 72 + kl;
                fb[nt].u[0] = *(const u32x4*)bp;       // K = kl+8h .. +7
                fb[nt].u[1] = *(const u32x4*)(bp + 16); // K = kl+16+8h .. +7
            }
#pragma unroll
            for (int nt = 0; nt < 4; ++nt) {
                acc[nt] = __builtin_amdgcn_wmma_f32_16x16x32_bf16(
                    false, fa.v, false, fb[nt].v, (short)0, acc[nt], false, false);
            }
        }
        __syncthreads();               // buffer free for next TDM overwrite
    }

    // epilogue
    int gm0 = bm * 128 + wav * 16;
    int gn0 = bn * 64;
#pragma unroll
    for (int nt = 0; nt < 4; ++nt) {
        int n = gn0 + nt * 16 + r;
        float bnv = bias[n];
#pragma unroll
        for (int i = 0; i < 8; ++i) {
            int m = gm0 + half * 8 + i;
            float val = acc[nt][i] + bnv;
            if (EPI == 1) {
                val = 0.5f * val * (1.0f + erff(val * 0.70710678f)); // exact GELU
            } else if (EPI == 2) {
                val += resid[(size_t)m * N + n];
            }
            size_t idx = (size_t)m * N + n;
            if (OUTB) Cb[idx] = f32_to_bf16(val);
            else      Cf[idx] = val;
        }
    }
}

// ---------- banded attention: one wave per (b,h,t) row, online softmax ----------
__global__ void k_banded_attn(const float* __restrict__ q,
                              const float* __restrict__ k,
                              const float* __restrict__ v,
                              unsigned short* __restrict__ ctx,
                              const int* __restrict__ band_ptr) {
    int wid  = (blockIdx.x * blockDim.x + threadIdx.x) >> 5;
    int lane = threadIdx.x & 31;
    int total = BB * HH * TT;
    if (wid >= total) return;
    int t  = wid % TT;
    int bh = wid / TT;
    int b  = bh / HH, h = bh % HH;

    size_t rowbase = ((size_t)(b * TT + t)) * DD + h * DHH;
    float q0 = q[rowbase + 2 * lane];
    float q1 = q[rowbase + 2 * lane + 1];

    int band = *band_ptr;
    int s0 = t - band; if (s0 < 0) s0 = 0;
    int s1 = t + band; if (s1 > TT - 1) s1 = TT - 1;

    const float scale = 0.125f;   // 1/sqrt(64)
    float m = -INFINITY, l = 0.f, a0 = 0.f, a1 = 0.f;
    for (int s = s0; s <= s1; ++s) {
        size_t kb = ((size_t)(b * TT + s)) * DD + h * DHH;
        float p = q0 * k[kb + 2 * lane] + q1 * k[kb + 2 * lane + 1];
#pragma unroll
        for (int off = 16; off; off >>= 1) p += __shfl_xor(p, off, 32);
        p *= scale;
        float mn = fmaxf(m, p);
        float c  = __expf(m - mn);
        float e  = __expf(p - mn);
        l  = l * c + e;
        a0 = a0 * c + e * v[kb + 2 * lane];
        a1 = a1 * c + e * v[kb + 2 * lane + 1];
        m = mn;
    }
    float inv = 1.0f / l;
    ctx[rowbase + 2 * lane]     = f32_to_bf16(a0 * inv);
    ctx[rowbase + 2 * lane + 1] = f32_to_bf16(a1 * inv);
}

// ---------- host ----------
template <int EPI, int OUTB>
static void launch_gemm(const unsigned short* A, const unsigned short* WT,
                        const float* bias, const float* resid,
                        float* Cf, unsigned short* Cb,
                        int M, int N, int K, hipStream_t s) {
    int blocks = (M / 128) * (N / 64);
    k_gemm<EPI, OUTB><<<dim3(blocks), dim3(256), 0, s>>>(A, WT, bias, resid, Cf, Cb, M, N, K);
}

extern "C" void kernel_launch(void* const* d_in, const int* in_sizes, int n_in,
                              void* d_out, int out_size, void* d_ws, size_t ws_size,
                              hipStream_t stream) {
    (void)in_sizes; (void)n_in; (void)out_size; (void)ws_size;
    const float* x_refined = (const float*)d_in[0];
    const float* x_mem     = (const float*)d_in[1];
    const float* ln_q_g    = (const float*)d_in[2];
    const float* ln_q_b    = (const float*)d_in[3];
    const float* ln_kv_g   = (const float*)d_in[4];
    const float* ln_kv_b   = (const float*)d_in[5];
    const float* Wq        = (const float*)d_in[6];
    const float* bq        = (const float*)d_in[7];
    const float* Wk        = (const float*)d_in[8];
    const float* bk        = (const float*)d_in[9];
    const float* Wv        = (const float*)d_in[10];
    const float* bv        = (const float*)d_in[11];
    const float* Wo        = (const float*)d_in[12];
    const float* bo        = (const float*)d_in[13];
    const float* ln_f_g    = (const float*)d_in[14];
    const float* ln_f_b    = (const float*)d_in[15];
    const float* W1        = (const float*)d_in[16];
    const float* b1        = (const float*)d_in[17];
    const float* W2        = (const float*)d_in[18];
    const float* b2        = (const float*)d_in[19];
    const int*   band      = (const int*)d_in[20];
    float* out = (float*)d_out;

    char* ws = (char*)d_ws;
    size_t off = 0;
    auto carve = [&](size_t bytes) -> char* {
        char* p = ws + off;
        off += (bytes + 255) & ~(size_t)255;
        return p;
    };
    const size_t MD = (size_t)MM * DD;
    unsigned short* Qn   = (unsigned short*)carve(MD * 2);
    unsigned short* KVn  = (unsigned short*)carve(MD * 2);
    unsigned short* WqT  = (unsigned short*)carve((size_t)DD * DD * 2);
    unsigned short* WkT  = (unsigned short*)carve((size_t)DD * DD * 2);
    unsigned short* WvT  = (unsigned short*)carve((size_t)DD * DD * 2);
    unsigned short* WoT  = (unsigned short*)carve((size_t)DD * DD * 2);
    unsigned short* W1T  = (unsigned short*)carve((size_t)DD * D4 * 2);
    unsigned short* W2T  = (unsigned short*)carve((size_t)DD * D4 * 2);
    float*          qf   = (float*)carve(MD * 4);
    float*          kf   = (float*)carve(MD * 4);
    float*          vf   = (float*)carve(MD * 4);
    unsigned short* ctxb = (unsigned short*)carve(MD * 2);
    float*          yf   = (float*)carve(MD * 4);
    unsigned short* hn   = (unsigned short*)carve(MD * 2);
    unsigned short* act  = (unsigned short*)carve((size_t)MM * D4 * 2);

    dim3 tb(32, 8);
    // weight transposes (fp32 -> bf16, W^T so WMMA B-operand reads are contiguous)
    k_transpose_bf16<<<dim3(DD / 32, DD / 32), tb, 0, stream>>>(Wq, WqT, DD, DD);
    k_transpose_bf16<<<dim3(DD / 32, DD / 32), tb, 0, stream>>>(Wk, WkT, DD, DD);
    k_transpose_bf16<<<dim3(DD / 32, DD / 32), tb, 0, stream>>>(Wv, WvT, DD, DD);
    k_transpose_bf16<<<dim3(DD / 32, DD / 32), tb, 0, stream>>>(Wo, WoT, DD, DD);
    k_transpose_bf16<<<dim3(D4 / 32, DD / 32), tb, 0, stream>>>(W1, W1T, DD, D4);
    k_transpose_bf16<<<dim3(DD / 32, D4 / 32), tb, 0, stream>>>(W2, W2T, D4, DD);

    // LayerNorms -> bf16 activations (wave per row; 4096 rows, 8 waves/block)
    k_layernorm_bf16<<<dim3(MM / 8), dim3(256), 0, stream>>>(x_refined, ln_q_g, ln_q_b, Qn, MM);
    k_layernorm_bf16<<<dim3(MM / 8), dim3(256), 0, stream>>>(x_mem, ln_kv_g, ln_kv_b, KVn, MM);

    // Q/K/V projections (bf16 WMMA + TDM staging, fp32 out)
    launch_gemm<0, 0>(Qn,  WqT, bq, nullptr, qf, nullptr, MM, DD, DD, stream);
    launch_gemm<0, 0>(KVn, WkT, bk, nullptr, kf, nullptr, MM, DD, DD, stream);
    launch_gemm<0, 0>(KVn, WvT, bv, nullptr, vf, nullptr, MM, DD, DD, stream);

    // banded attention: B*H*T = 65536 waves
    k_banded_attn<<<dim3((BB * HH * TT) / 8), dim3(256), 0, stream>>>(qf, kf, vf, ctxb, band);

    // O-projection + residual -> y (fp32)
    launch_gemm<2, 0>(ctxb, WoT, bo, x_refined, yf, nullptr, MM, DD, DD, stream);

    // final LN -> bf16
    k_layernorm_bf16<<<dim3(MM / 8), dim3(256), 0, stream>>>(yf, ln_f_g, ln_f_b, hn, MM);

    // FFN: GELU(h@W1+b1) (bf16 out), then @W2+b2 + y -> out (fp32)
    launch_gemm<1, 1>(hn,  W1T, b1, nullptr, nullptr, act, MM, D4, DD, stream);
    launch_gemm<2, 0>(act, W2T, b2, yf, out, nullptr, MM, DD, D4, stream);
}
